// SarvamMLAAttention_20392504721495
// MI455X (gfx1250) — compile-verified, matrix operand order checked by reading, not run on hardware
//
#include <hip/hip_runtime.h>
#include <hip/hip_bf16.h>
#include <stdint.h>

typedef __bf16 bf16;
typedef __attribute__((ext_vector_type(16))) __bf16 v16bf;
typedef __attribute__((ext_vector_type(8)))  float  v8f;

#define T_SEQ   2048
#define HIDDEN_ 4096
#define NH      32
#define DQK     192
#define DNOPE   128
#define DROPE   64
#define DV      128
#define QLORA   1536
#define KVLORA  512
#define KVA_N   576   /* KVLORA + DROPE */

// ---------------------------------------------------------------------------
// Fragment helpers (see round-1 comments for the 16-bit A/B VGPR layouts).
// Per-lane base = row_base + (lane>>4)*8; two contiguous 16B chunks at +0/+16.
// ---------------------------------------------------------------------------
__device__ __forceinline__ v16bf load_frag(const bf16* p) {
  union { v16bf v; uint4 q[2]; } f;
  f.q[0] = *(const uint4*)(p);
  f.q[1] = *(const uint4*)(p + 16);
  return f.v;
}

__device__ __forceinline__ v8f wmma_bf16f32(v16bf a, v16bf b, v8f c) {
  return __builtin_amdgcn_wmma_f32_16x16x32_bf16(false, a, false, b,
                                                 (short)0, c, false, false);
}

__device__ __forceinline__ void store_c(float* C, size_t i, float v) { C[i] = v; }
__device__ __forceinline__ void store_c(bf16*  C, size_t i, float v) { C[i] = (bf16)v; }

// ---------------------------------------------------------------------------
// CDNA5 async global->LDS (16B per lane), tracked by ASYNCcnt.
// LDS address operand: wave-relative LDS byte offset = low 32 bits of the
// generic pointer (ISA 10.2, LDS aperture).
// ---------------------------------------------------------------------------
__device__ __forceinline__ uint32_t lds_addr_of(const void* p) {
  return (uint32_t)(uintptr_t)p;
}
__device__ __forceinline__ void async_load_b128(uint32_t lds_dst, const bf16* gsrc) {
  asm volatile("global_load_async_to_lds_b128 %0, %1, off"
               :: "v"(lds_dst), "v"((uint64_t)(uintptr_t)gsrc) : "memory");
}
__device__ __forceinline__ void wait_asynccnt0() {
  asm volatile("s_wait_asynccnt 0x0" ::: "memory");
}

// ---------------------------------------------------------------------------
// f32 -> bf16 conversion (grid-stride)
// ---------------------------------------------------------------------------
__global__ void __launch_bounds__(256)
cvt_f32_bf16(const float* __restrict__ x, bf16* __restrict__ y, size_t n) {
  size_t i = (size_t)blockIdx.x * 256 + threadIdx.x;
  const size_t stride = (size_t)gridDim.x * 256;
  for (; i < n; i += stride) y[i] = (bf16)x[i];
}

// ---------------------------------------------------------------------------
// bf16 GEMM: C(MxN,row-major) = A(MxK,row-major,ld=lda) * B(KxN,row-major).
// Block tile 128 x BN, 8 waves, K-step 64, double-buffered LDS.
//   BN=128: waves 2x4, wave tile 64x32 (4x2 WMMA)
//   BN= 64: waves 4x2, wave tile 32x32 (2x2 WMMA)
// A tile staged with async global->LDS b128; B tile transposed to [n][k] by
// VALU so fragments are contiguous b128 LDS reads.  Next tile is staged while
// the current tile's 8..16 WMMAs execute; one s_wait_asynccnt + barrier per
// 64-deep K step.
// Requires: M%128==0, N%BN==0, K%64==0 (true for every call here).
// ---------------------------------------------------------------------------
template<typename OutT, int BN>
__global__ void __launch_bounds__(256)
gemm_bf16(const bf16* __restrict__ A, const bf16* __restrict__ B,
          OutT* __restrict__ C, int M, int N, int K, int lda) {
  constexpr int WCOLS = BN / 32;     // waves along N
  constexpr int WROWS = 8 / WCOLS;   // waves along M
  constexpr int MI    = 8 / WROWS;   // 16-row m-tiles per wave
  constexpr int NJ    = 2;           // 16-col n-tiles per wave
  __shared__ __align__(16) bf16 As[2][128 * 64];
  __shared__ __align__(16) bf16 BsT[2][BN * 64];   // [n][k]
  const int tid  = threadIdx.x;
  const int lane = tid & 31, wid = tid >> 5;
  const int half = lane >> 4, r = lane & 15;
  const int wrow = wid / WCOLS, wcol = wid % WCOLS;
  const int row0 = blockIdx.y * 128;
  const int col0 = blockIdx.x * BN;
  const int wm0  = wrow * (MI * 16);
  const int wn0  = wcol * 32;

  v8f zero = {};
  v8f acc[MI][NJ];
  for (int i = 0; i < MI; ++i)
    for (int j = 0; j < NJ; ++j) acc[i][j] = zero;

  auto stage_A = [&](int buf, int k0) {   // 128x64, 1024 b128 chunks, 4/thread
    for (int i = 0; i < 4; ++i) {
      int u = i * 256 + tid;
      int m = u >> 3, c8 = (u & 7) * 8;
      async_load_b128(lds_addr_of(&As[buf][m * 64 + c8]),
                      &A[(size_t)(row0 + m) * lda + k0 + c8]);
    }
  };
  auto stage_B = [&](int buf, int k0) {   // 64 x BN, transposed
    constexpr int DW = (64 * BN) / 2 / 256;
    for (int i = 0; i < DW; ++i) {
      int u = i * 256 + tid;
      int kk = u / (BN / 2), nn = (u % (BN / 2)) * 2;
      union { uint32_t w; bf16 h[2]; } cv;
      cv.w = *(const uint32_t*)&B[(size_t)(k0 + kk) * N + col0 + nn];
      BsT[buf][(nn + 0) * 64 + kk] = cv.h[0];
      BsT[buf][(nn + 1) * 64 + kk] = cv.h[1];
    }
  };

  stage_A(0, 0);
  stage_B(0, 0);
  wait_asynccnt0();
  __syncthreads();

  for (int k0 = 0; k0 < K; k0 += 64) {
    const int cur = (k0 >> 6) & 1, nxt = cur ^ 1;
    if (k0 + 64 < K) { stage_A(nxt, k0 + 64); stage_B(nxt, k0 + 64); }
    for (int ks = 0; ks < 2; ++ks) {
      v16bf afr[MI], bfr[NJ];
      for (int i = 0; i < MI; ++i)
        afr[i] = load_frag(&As[cur][(wm0 + i * 16 + r) * 64 + ks * 32 + half * 8]);
      for (int j = 0; j < NJ; ++j)
        bfr[j] = load_frag(&BsT[cur][(wn0 + j * 16 + r) * 64 + ks * 32 + half * 8]);
      for (int i = 0; i < MI; ++i)
        for (int j = 0; j < NJ; ++j)
          acc[i][j] = wmma_bf16f32(afr[i], bfr[j], acc[i][j]);
    }
    wait_asynccnt0();
    __syncthreads();
  }

  for (int i = 0; i < MI; ++i)
    for (int j = 0; j < NJ; ++j) {
      const int rr = row0 + wm0 + i * 16 + 8 * half;
      const int cc = col0 + wn0 + j * 16 + r;
      for (int s = 0; s < 8; ++s)
        store_c(C, (size_t)(rr + s) * N + cc, acc[i][j][s]);
    }
}

// ---------------------------------------------------------------------------
// In-place RMSNorm over first n columns of a row (row length ld), bf16.
// ---------------------------------------------------------------------------
__global__ void __launch_bounds__(256)
rmsnorm_inplace(bf16* __restrict__ x, const float* __restrict__ g, int n, int ld) {
  const int row = blockIdx.x;
  bf16* xr = x + (size_t)row * ld;
  __shared__ float red[256];
  float ss = 0.f;
  for (int i = threadIdx.x; i < n; i += 256) { float v = (float)xr[i]; ss += v * v; }
  red[threadIdx.x] = ss;
  __syncthreads();
  for (int s = 128; s > 0; s >>= 1) {
    if (threadIdx.x < s) red[threadIdx.x] += red[threadIdx.x + s];
    __syncthreads();
  }
  const float scale = rsqrtf(red[0] / (float)n + 1e-6f);
  for (int i = threadIdx.x; i < n; i += 256) {
    float v = (float)xr[i];
    xr[i] = (bf16)(v * scale * g[i]);
  }
}

// ---------------------------------------------------------------------------
// RoPE + head-major layout prep: Qf/Kf = [h][t][192] (Q scaled by 192^-0.5,
// interleaved RoPE on last 64 dims; k_pe broadcast), Vf = [h][t][128].
// ---------------------------------------------------------------------------
__global__ void __launch_bounds__(256)
rope_prep(const int* __restrict__ pos,
          const bf16* __restrict__ qproj,   // T x (NH*DQK)
          const bf16* __restrict__ kvproj,  // T x (NH*256)
          const bf16* __restrict__ kva,     // T x KVA_N (cols 512..575 = raw k_pe)
          bf16* __restrict__ Qf, bf16* __restrict__ Kf, bf16* __restrict__ Vf) {
  const int t = blockIdx.x;
  __shared__ float cs[32], sn[32], kpe[64];
  if (threadIdx.x < 32) {
    const float fp  = (float)pos[t];
    const float inv = __powf(10000.f, -((float)(2 * threadIdx.x)) / 64.f);
    const float a   = fp * inv;
    cs[threadIdx.x] = __cosf(a);
    sn[threadIdx.x] = __sinf(a);
  }
  __syncthreads();
  if (threadIdx.x < 32) {
    const float x1 = (float)kva[(size_t)t * KVA_N + KVLORA + 2 * threadIdx.x];
    const float x2 = (float)kva[(size_t)t * KVA_N + KVLORA + 2 * threadIdx.x + 1];
    kpe[2 * threadIdx.x]     = x1 * cs[threadIdx.x] - x2 * sn[threadIdx.x];
    kpe[2 * threadIdx.x + 1] = x1 * sn[threadIdx.x] + x2 * cs[threadIdx.x];
  }
  __syncthreads();
  const float scale = rsqrtf((float)DQK);
  for (int idx = threadIdx.x; idx < NH * DQK; idx += 256) {
    const int hh = idx / DQK, d = idx % DQK;
    float qv, kv;
    if (d < DNOPE) {
      qv = (float)qproj[(size_t)t * (NH * DQK) + hh * DQK + d];
      kv = (float)kvproj[(size_t)t * (NH * 256) + hh * 256 + d];
    } else {
      const int i = (d - DNOPE) >> 1;
      const size_t qb = (size_t)t * (NH * DQK) + hh * DQK + DNOPE + 2 * i;
      const float x1 = (float)qproj[qb];
      const float x2 = (float)qproj[qb + 1];
      qv = ((d & 1) == 0) ? (x1 * cs[i] - x2 * sn[i]) : (x1 * sn[i] + x2 * cs[i]);
      kv = kpe[d - DNOPE];
    }
    const size_t dst = ((size_t)hh * T_SEQ + t) * DQK + d;
    Qf[dst] = (bf16)(qv * scale);
    Kf[dst] = (bf16)kv;
  }
  for (int idx = threadIdx.x; idx < NH * DV; idx += 256) {
    const int hh = idx >> 7, d = idx & 127;
    Vf[((size_t)hh * T_SEQ + t) * DV + d] =
        kvproj[(size_t)t * (NH * 256) + hh * 256 + DNOPE + d];
  }
}

// ---------------------------------------------------------------------------
// Causal flash attention: one (head, 128-query block) per 256-thread block,
// each wave owns 16 query rows.  K tile (32x192) staged via async
// global->LDS; V tile transposed by VALU.  Online softmax in C layout,
// P re-laid-out to A fragments through a per-wave LDS patch.
// ---------------------------------------------------------------------------
__global__ void __launch_bounds__(256)
mla_attn(const bf16* __restrict__ Qf, const bf16* __restrict__ Kf,
         const bf16* __restrict__ Vf, bf16* __restrict__ O) {
  __shared__ __align__(16) bf16 Ks[32 * DQK];
  __shared__ __align__(16) bf16 VsT[DV * 32];     // [dv][key]
  __shared__ __align__(16) bf16 Ps[8][16 * 32];   // per-wave P staging
  const int h  = blockIdx.y;
  const int q0 = blockIdx.x * 128;
  const int tid  = threadIdx.x;
  const int lane = tid & 31, wid = tid >> 5;
  const int half = lane >> 4, r = lane & 15;
  const int qw = q0 + wid * 16;
  const bf16* Qh = Qf + (size_t)h * T_SEQ * DQK;
  const bf16* Kh = Kf + (size_t)h * T_SEQ * DQK;
  const bf16* Vh = Vf + (size_t)h * T_SEQ * DV;

  v16bf qfr[6];
  for (int c = 0; c < 6; ++c)
    qfr[c] = load_frag(Qh + (size_t)(qw + r) * DQK + c * 32 + half * 8);

  v8f zero = {};
  v8f oacc[8];
  for (int c = 0; c < 8; ++c) oacc[c] = zero;
  float mrow[8], lrow[8];
  for (int s = 0; s < 8; ++s) { mrow[s] = -3.0e38f; lrow[s] = 0.f; }

  const int kend = q0 + 128;
  for (int k0 = 0; k0 < kend; k0 += 32) {
    // K tile 32x192 = 768 b128 chunks, 3 per thread, async global->LDS
    for (int i = 0; i < 3; ++i) {
      int u = i * 256 + tid;
      int kk = u / 24, c8 = (u % 24) * 8;
      async_load_b128(lds_addr_of(&Ks[kk * DQK + c8]),
                      &Kh[(size_t)(k0 + kk) * DQK + c8]);
    }
    // V tile transposed (dv x 32 keys)
    for (int i = 0; i < 8; ++i) {
      int u = i * 256 + tid;
      int kk = u >> 6, dd = (u & 63) * 2;
      union { uint32_t w; bf16 h2[2]; } cv;
      cv.w = *(const uint32_t*)&Vh[(size_t)(k0 + kk) * DV + dd];
      VsT[(dd + 0) * 32 + kk] = cv.h2[0];
      VsT[(dd + 1) * 32 + kk] = cv.h2[1];
    }
    wait_asynccnt0();
    __syncthreads();

    // S = Q * K^T : two 16-key column tiles, 6 K-steps each
    v8f sacc[2];
    for (int j = 0; j < 2; ++j) {
      sacc[j] = zero;
      for (int c = 0; c < 6; ++c) {
        v16bf kfr = load_frag(&Ks[(j * 16 + r) * DQK + c * 32 + half * 8]);
        sacc[j] = wmma_bf16f32(qfr[c], kfr, sacc[j]);
      }
    }

    // causal mask + online softmax (C layout: col = r, row = slot + 8*half)
    float pm[2][8], mnew[8];
    for (int s = 0; s < 8; ++s) {
      const int qrow = qw + s + 8 * half;
      float mx = mrow[s];
      for (int j = 0; j < 2; ++j) {
        const int key = k0 + j * 16 + r;
        float v = sacc[j][s];
        if (key > qrow) v = -3.0e38f;
        pm[j][s] = v;
        mx = fmaxf(mx, v);
      }
      for (int m = 1; m < 16; m <<= 1) mx = fmaxf(mx, __shfl_xor(mx, m, 32));
      mnew[s] = mx;
    }
    for (int s = 0; s < 8; ++s) {
      const float alpha = __expf(mrow[s] - mnew[s]);
      float lsum = 0.f;
      for (int j = 0; j < 2; ++j) {
        const float pv = __expf(pm[j][s] - mnew[s]);
        pm[j][s] = pv;
        lsum += pv;
      }
      for (int m = 1; m < 16; m <<= 1) lsum += __shfl_xor(lsum, m, 32);
      lrow[s] = lrow[s] * alpha + lsum;
      mrow[s] = mnew[s];
      for (int c = 0; c < 8; ++c) oacc[c][s] *= alpha;
    }

    // P (16x32) C-layout -> A-fragment via per-wave LDS patch
    for (int s = 0; s < 8; ++s)
      for (int j = 0; j < 2; ++j)
        Ps[wid][(s + 8 * half) * 32 + j * 16 + r] = (bf16)pm[j][s];
    asm volatile("s_wait_dscnt 0x0" ::: "memory");
    v16bf pfr = load_frag(&Ps[wid][r * 32 + half * 8]);

    // O += P * V  (8 dv column tiles)
    for (int c = 0; c < 8; ++c) {
      v16bf vfr = load_frag(&VsT[(c * 16 + r) * 32 + half * 8]);
      oacc[c] = wmma_bf16f32(pfr, vfr, oacc[c]);
    }
    __syncthreads();
  }

  for (int s = 0; s < 8; ++s) {
    const float inv = 1.f / lrow[s];
    const int qrow = qw + s + 8 * half;
    for (int c = 0; c < 8; ++c)
      O[(size_t)qrow * (NH * DV) + h * DV + c * 16 + r] = (bf16)(oacc[c][s] * inv);
  }
}

// ---------------------------------------------------------------------------
// Host-side orchestration
// ---------------------------------------------------------------------------
extern "C" void kernel_launch(void* const* d_in, const int* in_sizes, int n_in,
                              void* d_out, int out_size, void* d_ws, size_t ws_size,
                              hipStream_t stream) {
  (void)in_sizes; (void)n_in; (void)out_size; (void)ws_size;
  const int*   pos    = (const int*)  d_in[0];
  const float* hidden = (const float*)d_in[1];
  const float* wq_a   = (const float*)d_in[2];
  const float* gq     = (const float*)d_in[3];
  const float* wq_b   = (const float*)d_in[4];
  const float* wkv_a  = (const float*)d_in[5];
  const float* gkv    = (const float*)d_in[6];
  const float* wkv_b  = (const float*)d_in[7];
  const float* wo     = (const float*)d_in[8];
  float* out = (float*)d_out;

  char* p = (char*)d_ws;
  auto alloc_bf = [&](size_t elems) -> bf16* {
    bf16* r = (bf16*)p;
    p += ((elems * sizeof(bf16)) + 255) & ~(size_t)255;
    return r;
  };
  bf16* hid_bf  = alloc_bf((size_t)T_SEQ * HIDDEN_);
  bf16* wqa_bf  = alloc_bf((size_t)HIDDEN_ * QLORA);
  bf16* wkva_bf = alloc_bf((size_t)HIDDEN_ * KVA_N);
  bf16* wqb_bf  = alloc_bf((size_t)QLORA * (NH * DQK));
  bf16* wkvb_bf = alloc_bf((size_t)KVLORA * (NH * 256));
  bf16* wo_bf   = alloc_bf((size_t)(NH * DV) * HIDDEN_);
  bf16* qlora   = alloc_bf((size_t)T_SEQ * QLORA);
  bf16* kva     = alloc_bf((size_t)T_SEQ * KVA_N);
  bf16* qproj   = alloc_bf((size_t)T_SEQ * NH * DQK);
  bf16* kvproj  = alloc_bf((size_t)T_SEQ * NH * 256);
  bf16* Qh      = alloc_bf((size_t)NH * T_SEQ * DQK);
  bf16* Kh      = alloc_bf((size_t)NH * T_SEQ * DQK);
  bf16* Vh      = alloc_bf((size_t)NH * T_SEQ * DV);
  bf16* attn    = alloc_bf((size_t)T_SEQ * NH * DV);

  auto cvt = [&](const float* src, bf16* dst, size_t n) {
    cvt_f32_bf16<<<2048, 256, 0, stream>>>(src, dst, n);
  };
  cvt(hidden, hid_bf,  (size_t)T_SEQ * HIDDEN_);
  cvt(wq_a,   wqa_bf,  (size_t)HIDDEN_ * QLORA);
  cvt(wkv_a,  wkva_bf, (size_t)HIDDEN_ * KVA_N);
  cvt(wq_b,   wqb_bf,  (size_t)QLORA * NH * DQK);
  cvt(wkv_b,  wkvb_bf, (size_t)KVLORA * NH * 256);
  cvt(wo,     wo_bf,   (size_t)(NH * DV) * HIDDEN_);

  // q_lora = hidden @ wq_a ; kv_a = hidden @ wkv_a
  gemm_bf16<bf16, 128><<<dim3(QLORA / 128, T_SEQ / 128), 256, 0, stream>>>(
      hid_bf, wqa_bf, qlora, T_SEQ, QLORA, HIDDEN_, HIDDEN_);
  gemm_bf16<bf16, 64><<<dim3(KVA_N / 64, T_SEQ / 128), 256, 0, stream>>>(
      hid_bf, wkva_bf, kva, T_SEQ, KVA_N, HIDDEN_, HIDDEN_);

  // rmsnorms (in place; k_pe columns of kva stay raw)
  rmsnorm_inplace<<<T_SEQ, 256, 0, stream>>>(qlora, gq, QLORA, QLORA);
  rmsnorm_inplace<<<T_SEQ, 256, 0, stream>>>(kva, gkv, KVLORA, KVA_N);

  // q = qn @ wq_b ; kv = ckvn @ wkv_b
  gemm_bf16<bf16, 128><<<dim3((NH * DQK) / 128, T_SEQ / 128), 256, 0, stream>>>(
      qlora, wqb_bf, qproj, T_SEQ, NH * DQK, QLORA, QLORA);
  gemm_bf16<bf16, 128><<<dim3((NH * 256) / 128, T_SEQ / 128), 256, 0, stream>>>(
      kva, wkvb_bf, kvproj, T_SEQ, NH * 256, KVLORA, KVA_N);

  // RoPE + head-major Q/K/V
  rope_prep<<<T_SEQ, 256, 0, stream>>>(pos, qproj, kvproj, kva, Qh, Kh, Vh);

  // causal flash attention
  mla_attn<<<dim3(T_SEQ / 128, NH), 256, 0, stream>>>(Qh, Kh, Vh, attn);

  // out = attn @ wo (f32 output)
  gemm_bf16<float, 128><<<dim3(HIDDEN_ / 128, T_SEQ / 128), 256, 0, stream>>>(
      attn, wo_bf, out, T_SEQ, HIDDEN_, NH * DV, NH * DV);
}
